// UniMPTransformer_18073222382227
// MI455X (gfx1250) — matrix operand order repaired and checked by reading, not compiled, not run on hardware
//
#include <hip/hip_runtime.h>
#include <hip/hip_bf16.h>
#include <math.h>

typedef __attribute__((ext_vector_type(16))) _Float16 v16h;
typedef __attribute__((ext_vector_type(8)))  _Float16 v8h;
typedef __attribute__((ext_vector_type(8)))  float    v8f;

#define EMB 128
#define NH  4
#define DH  32
#define GN_EPS 1e-5f

// ---- monotone float <-> uint encoding for atomicMax-based segment max ----
__device__ __forceinline__ unsigned fenc(float f) {
    unsigned u = __float_as_uint(f);
    return (u & 0x80000000u) ? ~u : (u | 0x80000000u);
}
__device__ __forceinline__ float fdec(unsigned k) {
    unsigned u = (k & 0x80000000u) ? (k ^ 0x80000000u) : ~k;
    return __uint_as_float(u);
}

// ---- convert a 128x128 fp32 weight [fin][fout] to transposed f16 [fout][fin] ----
__global__ void k_cvt_w(const float* __restrict__ W, _Float16* __restrict__ Wt) {
    int i  = blockIdx.x * blockDim.x + threadIdx.x;   // 0..16383
    int fo = i >> 7;
    int fi = i & 127;
    Wt[fo * EMB + fi] = (_Float16)W[fi * EMB + fo];
}

// ---- fp32 -> f16 elementwise ----
__global__ void k_cvt_x(const float* __restrict__ x, _Float16* __restrict__ xh, int n) {
    int i = blockIdx.x * blockDim.x + threadIdx.x;
    if (i < n) xh[i] = (_Float16)x[i];
}

// ---- WMMA GEMM: C[M x 128] = A16[M x 128] @ Wt^T + bias, optional ReLU ----
// One wave computes a 16-row x 128-col stripe: 8 accumulator tiles, K=128 in 4 steps.
__global__ void k_gemm(const _Float16* __restrict__ A,
                       const _Float16* __restrict__ Wt,   // [128 (out)][128 (in)] f16
                       const float*    __restrict__ bias, // [128]
                       float*          __restrict__ C,
                       int M, int relu) {
    int lane = threadIdx.x & 31;
    int wid  = threadIdx.x >> 5;
    int tile = blockIdx.x * (blockDim.x >> 5) + wid;
    int ntiles = (M + 15) >> 4;
    if (tile >= ntiles) return;

    int r0  = tile << 4;
    int mlo = lane & 15;       // row (A) / col (B,D) within tile
    int hi  = lane >> 4;       // lane-half selector

    int arow = r0 + mlo;
    if (arow >= M) arow = M - 1;
    const _Float16* arp = A + (size_t)arow * EMB;

    v8f acc[8] = {};

    #pragma unroll
    for (int kb = 0; kb < 4; ++kb) {
        // A fragment (16-bit A 16x32 layout): elements 0..7 -> K = kb*32 + hi*8 + e,
        //                                     elements 8..15 -> K = kb*32 + 16 + hi*8 + e
        v8h alo = *(const v8h*)(arp + kb * 32 + hi * 8);
        v8h ahi = *(const v8h*)(arp + kb * 32 + 16 + hi * 8);
        v16h a;
        #pragma unroll
        for (int i = 0; i < 8; ++i) { a[i] = alo[i]; a[8 + i] = ahi[i]; }

        #pragma unroll
        for (int nt = 0; nt < 8; ++nt) {
            // B fragment (32x16): lane holds column n = nt*16+mlo, same K order as A (kb*32 + hi*8 + e, kb*32 + 16 + hi*8 + e)
            const _Float16* bp = Wt + (size_t)(nt * 16 + mlo) * EMB + kb * 32 + hi * 8;
            v8h blo = *(const v8h*)(bp);
            v8h bhi = *(const v8h*)(bp + 16);
            v16h b;
            #pragma unroll
            for (int i = 0; i < 8; ++i) { b[i] = blo[i]; b[8 + i] = bhi[i]; }
            acc[nt] = __builtin_amdgcn_wmma_f32_16x16x32_f16(
                false, a, false, b, (short)0, acc[nt], false, false);
        }
    }

    #pragma unroll
    for (int nt = 0; nt < 8; ++nt) {
        int col = nt * 16 + mlo;
        float bb = bias[col];
        #pragma unroll
        for (int v = 0; v < 8; ++v) {
            int row = r0 + v + hi * 8;   // D layout: VGPR v -> M = v + 8*hi, N = lane&15
            if (row < M) {
                float val = acc[nt][v] + bb;
                if (relu) val = fmaxf(val, 0.f);
                C[(size_t)row * EMB + col] = val;
            }
        }
    }
}

// ---- per-edge attention logits + segment max ----
__global__ void k_logits(const int* __restrict__ ei, const float* __restrict__ Q,
                         const float* __restrict__ K, float* __restrict__ elog,
                         unsigned* __restrict__ mmax, int E) {
    int idx = blockIdx.x * blockDim.x + threadIdx.x;
    if (idx >= E * NH) return;
    int e = idx >> 2, h = idx & 3;
    int s = ei[e], d = ei[E + e];
    const float4* qp = (const float4*)(Q + (size_t)d * EMB + h * DH);
    const float4* kp = (const float4*)(K + (size_t)s * EMB + h * DH);
    float acc = 0.f;
    #pragma unroll
    for (int i = 0; i < 8; ++i) {
        float4 qa = qp[i], ka = kp[i];
        acc += qa.x * ka.x + qa.y * ka.y + qa.z * ka.z + qa.w * ka.w;
    }
    acc *= 0.17677669529663687f;   // 1/sqrt(DH)
    elog[idx] = acc;
    atomicMax(&mmax[d * NH + h], fenc(acc));
}

// ---- exp(logit - max) + segment sum ----
__global__ void k_expsum(const int* __restrict__ ei, float* __restrict__ elog,
                         const unsigned* __restrict__ mmax, float* __restrict__ zsum, int E) {
    int idx = blockIdx.x * blockDim.x + threadIdx.x;
    if (idx >= E * NH) return;
    int e = idx >> 2, h = idx & 3;
    int d = ei[E + e];
    float v = expf(elog[idx] - fdec(mmax[d * NH + h]));
    elog[idx] = v;
    atomicAdd(&zsum[d * NH + h], v);
}

// ---- scatter alpha * v[src] into AGG[dst] (AGG pre-initialized with skip GEMM) ----
__global__ void k_scatter(const int* __restrict__ ei, const float* __restrict__ elog,
                          const float* __restrict__ zsum, const float* __restrict__ V,
                          float* __restrict__ AGG, int E) {
    int idx = blockIdx.x * blockDim.x + threadIdx.x;
    if (idx >= E * NH) return;
    int e = idx >> 2, h = idx & 3;
    int s = ei[e], d = ei[E + e];
    float alpha = elog[idx] / zsum[d * NH + h];
    const float* vp = V + (size_t)s * EMB + h * DH;
    float* op = AGG + (size_t)d * EMB + h * DH;
    #pragma unroll
    for (int i = 0; i < DH; ++i) atomicAdd(&op[i], alpha * vp[i]);
}

// ---- per-column sum & sumsq (coalesced: thread == column, block strides rows) ----
__global__ void k_colstat(const float* __restrict__ AGG, float* __restrict__ cs, int M) {
    int c = threadIdx.x;  // 0..127
    float s = 0.f, s2 = 0.f;
    for (int r = blockIdx.x; r < M; r += gridDim.x) {
        float v = AGG[(size_t)r * EMB + c];
        s += v; s2 += v * v;
    }
    atomicAdd(&cs[c], s);
    atomicAdd(&cs[EMB + c], s2);
}

// ---- GraphNorm + ReLU, emit f16 activations for next GEMM stage ----
__global__ void k_norm(const float* __restrict__ AGG, const float* __restrict__ cs,
                       const float* __restrict__ w, const float* __restrict__ b,
                       const float* __restrict__ ms, _Float16* __restrict__ out16, int M) {
    int idx = blockIdx.x * blockDim.x + threadIdx.x;
    if (idx >= M * EMB) return;
    int c = idx & 127;
    float inv  = 1.f / (float)M;
    float mean = cs[c] * inv;
    float ex2  = cs[EMB + c] * inv;
    float mm   = ms[c] * mean;
    float var  = ex2 - 2.f * mm * mean + mm * mm;   // E[(x - ms*mean)^2]
    float xs   = AGG[idx] - mm;
    float y    = w[c] * xs * rsqrtf(var + GN_EPS) + b[c];
    out16[idx] = (_Float16)fmaxf(y, 0.f);
}

// ---- final 128 -> 2 head ----
__global__ void k_final(const float* __restrict__ Hb, const float* __restrict__ lw1,
                        const float* __restrict__ lb1, float* __restrict__ out, int M) {
    int n = blockIdx.x * blockDim.x + threadIdx.x;
    if (n >= M) return;
    float a0 = lb1[0], a1 = lb1[1];
    const float* hp = Hb + (size_t)n * EMB;
    #pragma unroll 16
    for (int i = 0; i < EMB; ++i) {
        float h = hp[i];
        a0 += h * lw1[2 * i];
        a1 += h * lw1[2 * i + 1];
    }
    out[2 * n]     = a0;
    out[2 * n + 1] = a1;
}

extern "C" void kernel_launch(void* const* d_in, const int* in_sizes, int n_in,
                              void* d_out, int out_size, void* d_ws, size_t ws_size,
                              hipStream_t stream) {
    const float* x  = (const float*)d_in[0];
    const int*   ei = (const int*)d_in[1];
    int Nn = in_sizes[0] / EMB;
    int Ee = in_sizes[1] / 2;

    // input dict order: x, edge_index, then per layer {Wq,bq,Wk,bk,Wv,bv,Ws,bs,gnw,gnb,gnm}, lw0,lb0,lw1,lb1
    const float* Wsrc[9] = {
        (const float*)d_in[2],  (const float*)d_in[4],  (const float*)d_in[6],  (const float*)d_in[8],   // q,k,v,s layer0
        (const float*)d_in[13], (const float*)d_in[15], (const float*)d_in[17], (const float*)d_in[19],  // q,k,v,s layer1
        (const float*)d_in[24]                                                                            // lw0
    };
    const float* bq[2]  = {(const float*)d_in[3],  (const float*)d_in[14]};
    const float* bk[2]  = {(const float*)d_in[5],  (const float*)d_in[16]};
    const float* bv[2]  = {(const float*)d_in[7],  (const float*)d_in[18]};
    const float* bs[2]  = {(const float*)d_in[9],  (const float*)d_in[20]};
    const float* gnw[2] = {(const float*)d_in[10], (const float*)d_in[21]};
    const float* gnb[2] = {(const float*)d_in[11], (const float*)d_in[22]};
    const float* gnm[2] = {(const float*)d_in[12], (const float*)d_in[23]};
    const float* lb0 = (const float*)d_in[25];
    const float* lw1 = (const float*)d_in[26];
    const float* lb1 = (const float*)d_in[27];

    // ---- carve workspace ----
    char* base = (char*)d_ws;
    size_t off = 0;
    auto carve = [&](size_t bytes) -> void* {
        void* p = base + off;
        off = (off + bytes + 255) & ~(size_t)255;
        return p;
    };
    _Float16* A16  = (_Float16*)carve((size_t)Nn * EMB * sizeof(_Float16));
    float*    Q    = (float*)   carve((size_t)Nn * EMB * sizeof(float));
    float*    Kb   = (float*)   carve((size_t)Nn * EMB * sizeof(float));
    float*    Vb   = (float*)   carve((size_t)Nn * EMB * sizeof(float));
    float*    AGG  = (float*)   carve((size_t)Nn * EMB * sizeof(float));
    float*    ELOG = (float*)   carve((size_t)Ee * NH  * sizeof(float));
    unsigned* MMAX = (unsigned*)carve((size_t)Nn * NH  * sizeof(unsigned));
    float*    ZSUM = (float*)   carve((size_t)Nn * NH  * sizeof(float));
    float*    CS   = (float*)   carve(2 * EMB * sizeof(float));
    _Float16* WT[9];
    for (int i = 0; i < 9; ++i)
        WT[i] = (_Float16*)carve((size_t)EMB * EMB * sizeof(_Float16));

    // ---- weight transpose+convert, activation convert ----
    for (int i = 0; i < 9; ++i)
        k_cvt_w<<<64, 256, 0, stream>>>(Wsrc[i], WT[i]);
    k_cvt_x<<<(Nn * EMB + 255) / 256, 256, 0, stream>>>(x, A16, Nn * EMB);

    int tiles  = (Nn + 15) >> 4;
    int ggrid  = (tiles + 3) / 4;                 // 4 waves per block
    int eth    = Ee * NH;
    int egrid  = (eth + 255) / 256;
    int ngrid  = (Nn * EMB + 255) / 256;

    for (int l = 0; l < 2; ++l) {
        int wb = l * 4;
        // skip connection directly into AGG; q/k/v projections
        k_gemm<<<ggrid, 128, 0, stream>>>(A16, WT[wb + 3], bs[l], AGG, Nn, 0);
        k_gemm<<<ggrid, 128, 0, stream>>>(A16, WT[wb + 0], bq[l], Q,   Nn, 0);
        k_gemm<<<ggrid, 128, 0, stream>>>(A16, WT[wb + 1], bk[l], Kb,  Nn, 0);
        k_gemm<<<ggrid, 128, 0, stream>>>(A16, WT[wb + 2], bv[l], Vb,  Nn, 0);

        hipMemsetAsync(MMAX, 0, (size_t)Nn * NH * sizeof(unsigned), stream); // fenc(-inf) < any real key
        hipMemsetAsync(ZSUM, 0, (size_t)Nn * NH * sizeof(float), stream);
        hipMemsetAsync(CS,   0, 2 * EMB * sizeof(float), stream);

        k_logits <<<egrid, 256, 0, stream>>>(ei, Q, Kb, ELOG, MMAX, Ee);
        k_expsum <<<egrid, 256, 0, stream>>>(ei, ELOG, MMAX, ZSUM, Ee);
        k_scatter<<<egrid, 256, 0, stream>>>(ei, ELOG, ZSUM, Vb, AGG, Ee);

        k_colstat<<<256, 128, 0, stream>>>(AGG, CS, Nn);
        k_norm<<<ngrid, 256, 0, stream>>>(AGG, CS, gnw[l], gnb[l], gnm[l], A16, Nn);
    }

    // MLP: relu(h @ lw0 + lb0) via WMMA, then 128->2 head
    k_gemm<<<ggrid, 128, 0, stream>>>(A16, WT[8], lb0, Q, Nn, 1);
    k_final<<<(Nn + 255) / 256, 256, 0, stream>>>(Q, lw1, lb1, (float*)d_out, Nn);
}
